// SemiMarkovCRF_88338887344543
// MI455X (gfx1250) — matrix operand order repaired and checked by reading, not compiled, not run on hardware
//
#include <hip/hip_runtime.h>
#include <hip/hip_bf16.h>

// Semi-Markov CRF forward logZ for MI455X (gfx1250, wave32).
//
// B=512, P=24, L=288, DMAX=96.  One workgroup handles a tile of 16 batch
// elements; the whole DP state lives in LDS (CDNA5: 320KB/WGP):
//   Q[slot][b][p] = beta[tau] - Ucp[b,p,tau+1]   (96x16x24 f32 ring, 147KB)
// so that  cand(d) = Q[tau] + Ucp[t+1] + dur[p][d-1]  and the constant
// Ucp[t+1] factors out of the duration logsumexp.
// The Potts transition  beta = LSE_k(alpha[k] + A[k,p'])  is a log-space
// GEMM:  beta = M_b + log( exp(alpha - M_b) @ W ),  W = I + exp(-eta)(1-I),
// executed with v_wmma_f32_16x16x32_f16 (M=16 batches, K/N padded to 32).
// theta is streamed into a double-buffered LDS chunk by the Tensor Data
// Mover (tensor_load_to_lds, 6-arg clang-23 form), sync'd with
// s_wait_tensorcnt.

#define B_TOT 512
#define P_N   24
#define L_N   288
#define DMAXC 96
#define ETA_C 4.0f
#define TB    16          // batch tile per workgroup
#define CHN   32          // theta steps per LDS chunk (288 % 32 == 0)
#define NEG_INF (-8.507059173023462e37f)   // finfo(f32).min / 4

typedef __attribute__((ext_vector_type(16))) _Float16     v16h;
typedef __attribute__((ext_vector_type(8)))  float        v8f;
typedef __attribute__((ext_vector_type(4)))  unsigned int v4u;
typedef __attribute__((ext_vector_type(8)))  int          v8i;
typedef __attribute__((ext_vector_type(4)))  int          v4i;

__global__ __launch_bounds__(256)
void smcrf_fwd_kernel(const float* __restrict__ theta,        // [B,P,L]
                      const float* __restrict__ dur,          // [P,DMAX]
                      const unsigned char* __restrict__ mask, // [L,P] bool
                      float* __restrict__ out)                // [B]
{
    __shared__ float         Q[DMAXC * TB * P_N];      // 147456 B ring buffer
    __shared__ float         thetaC[2][TB * P_N * CHN];// 2 x 49152 B chunks
    __shared__ float         durS[P_N * DMAXC];        // 9216 B
    __shared__ unsigned char allowS[L_N * P_N];        // 6912 B
    __shared__ float         US[TB * P_N];             // running cumsum Ucp[t+1]
    __shared__ float         alphaS[TB * P_N];
    __shared__ float         MS[TB];                   // per-batch row max
    __shared__ _Float16      ES[TB * 32];              // exp(alpha-M), K-padded
    __shared__ _Float16      WS[32 * 32];              // Potts exp-matrix, padded

    const int tid  = threadIdx.x;
    const int lane = tid & 31;
    const int wave = tid >> 5;
    const int hi   = lane >> 4;
    const int b0   = blockIdx.x * TB;

    // ---- TDM: one 2D descriptor moves a whole [384 rows x 32 floats] chunk.
    // Rows are (b,p) pairs: global row index (b0*24 + i), row stride L floats,
    // rows land contiguously in LDS == thetaC[buf] layout [TB*P_N][CHN].
    auto issue_tdm = [&](int tstart, int buf) {
        unsigned long long ga =
            (unsigned long long)(const void*)theta +
            ((unsigned long long)b0 * P_N * L_N + (unsigned long long)tstart) * 4ull;
        unsigned ldsa = (unsigned)(unsigned long long)(uintptr_t)&thetaC[buf][0];
        v4u g0;
        g0[0] = 1u;                                    // count=1, user descriptor
        g0[1] = ldsa;                                  // lds_addr (bytes)
        g0[2] = (unsigned)(ga & 0xffffffffull);        // global_addr[31:0]
        g0[3] = (unsigned)((ga >> 32) & 0x01ffffffull) // global_addr[56:32]
              | (2u << 30);                            // type = 2 ("image")
        v8i g1;
        g1[0] = (int)(2u << 16);                       // data_size = 4 bytes
        g1[1] = (int)((unsigned)L_N << 16);            // tensor_dim0[15:0] @ [63:48]
        g1[2] = (int)((unsigned)(TB * P_N) << 16);     // dim0 hi=0 | tensor_dim1 lo16
        g1[3] = (int)((unsigned)CHN << 16);            // dim1 hi=0 | tile_dim0 = 32
        g1[4] = (int)(TB * P_N);                       // tile_dim1 = 384, tile_dim2=0
        g1[5] = (int)L_N;                              // tensor_dim0_stride = 288
        g1[6] = 0;
        g1[7] = 0;                                     // tensor_dim1_stride unused (2D)
        v4i z4 = {0, 0, 0, 0};
        v8i z8 = {0, 0, 0, 0, 0, 0, 0, 0};
        // clang-23 / therock-10.0 6-arg form:
        // (g0, g1, g2, g3, g4, cachepolicy)
        __builtin_amdgcn_tensor_load_to_lds(g0, g1, z4, z4, z8, 0);
    };

    // Kick off chunk 0 into buffer 0 immediately (overlaps with LDS init).
    if (wave == 0) issue_tdm(0, 0);

    // ---------------- init ----------------
    for (int i = tid; i < DMAXC * TB * P_N; i += 256) Q[i] = NEG_INF;
    for (int i = tid; i < P_N * DMAXC;      i += 256) durS[i] = dur[i];
    for (int i = tid; i < L_N * P_N;        i += 256) allowS[i] = mask[i];
    for (int i = tid; i < TB * P_N;         i += 256) US[i] = 0.0f;
    for (int i = tid; i < 32 * 32;          i += 256) {
        int k = i >> 5, n = i & 31;
        float w = (k < P_N && n < P_N) ? ((k == n) ? 1.0f : __expf(-ETA_C)) : 0.0f;
        WS[i] = (_Float16)w;
    }
    __syncthreads();
    // tau = -1 lives in slot 95: beta(-1)=0, Ucp[...,0]=0  ->  Q = 0
    for (int i = tid; i < TB * P_N; i += 256) Q[(DMAXC - 1) * TB * P_N + i] = 0.0f;
    __syncthreads();

    // ---------------- sequential scan over t ----------------
    int curbuf = 0;
    for (int t = 0; t < L_N; ++t) {
        const int tt = t & (CHN - 1);
        if (tt == 0) {
            curbuf = (t >> 5) & 1;
            if (wave == 0) __builtin_amdgcn_s_wait_tensorcnt(0); // chunk ready
            __syncthreads();                                     // publish LDS
            if (wave == 0 && (t + CHN) < L_N)
                issue_tdm(t + CHN, curbuf ^ 1);                  // prefetch next
        }
        const float* __restrict__ thC = thetaC[curbuf];

        // phase 1: running cumsum  US = Ucp[...,t+1]
        for (int pair = tid; pair < TB * P_N; pair += 256)
            US[pair] += thC[pair * CHN + tt];
        __syncthreads();

        // phase 2: duration logsumexp -> alpha[b][p]
        for (int pair = tid; pair < TB * P_N; pair += 256) {
            const int b = pair / P_N, p = pair - b * P_N;
            float m = NEG_INF, s = 0.0f;
            if (allowS[t * P_N + p]) {                 // segment-end mask
                const int jmax = (t < DMAXC - 1) ? t : (DMAXC - 1);
                const float* __restrict__ durp = durS + p * DMAXC;
                const float* __restrict__ Qbp  = Q + b * P_N + p;
                int slot = t > 0 ? (t - 1) % DMAXC : DMAXC - 1;  // slot for j=0
                int aidx = t * P_N + p;                          // allow[t-j][p]
                for (int j = 0; j <= jmax; ++j) {                // j = d-1
                    if (allowS[aidx]) {                          // segment-start
                        const float v = Qbp[slot * (TB * P_N)] + durp[j];
                        if (v <= m) { s += __expf(v - m); }
                        else        { s = s * __expf(m - v) + 1.0f; m = v; }
                    }
                    aidx -= P_N;                                 // t-j decrements
                    slot = slot ? slot - 1 : DMAXC - 1;          // ring decrement
                }
            }
            alphaS[pair] = (s > 0.0f) ? (US[pair] + m + __logf(s)) : NEG_INF;
        }
        __syncthreads();

        // phase 3: per-batch row max; final answer at t = L-1
        if (tid < TB) {
            float m = NEG_INF;
            for (int p = 0; p < P_N; ++p) m = fmaxf(m, alphaS[tid * P_N + p]);
            MS[tid] = m;
            if (t == L_N - 1) {
                float s = 0.0f;
                for (int p = 0; p < P_N; ++p) s += __expf(alphaS[tid * P_N + p] - m);
                out[b0 + tid] = m + __logf(s);
            }
        }
        __syncthreads();

        // phase 4: E[b][k] = exp(alpha - M), zero-padded K to 32
        for (int i = tid; i < TB * 32; i += 256) {
            int b = i >> 5, k = i & 31;
            float e = (k < P_N) ? __expf(alphaS[b * P_N + k] - MS[b]) : 0.0f;
            ES[i] = (_Float16)e;
        }
        __syncthreads();

        // phase 5: Potts transition as log-space GEMM via WMMA.
        // waves 0/1 each own a 16-wide N tile; D[b][p'] = sum_k E[b][k]*W[k][p']
        if (wave < 2) {
            const int n0 = wave * 16;
            const int mr = lane & 15;
            v16h af, bf;
            // 16-bit A 16x32 striping as half index e:
            //   k = e + (e>=8 ? 8 : 0) + (hi ? 8 : 0)
            #pragma unroll
            for (int e = 0; e < 16; ++e) {
                const int ka = e + ((e >= 8) ? 8 : 0) + (hi ? 8 : 0);
                af[e] = ES[mr * 32 + ka];
                const int kb = e + ((e >= 8) ? 8 : 0) + (hi ? 8 : 0);   // B 32x16: same K striping as A
                bf[e] = WS[kb * 32 + n0 + mr];
            }
            v8f c = {};
            c = __builtin_amdgcn_wmma_f32_16x16x32_f16(
                    /*neg_a=*/false, af, /*neg_b=*/false, bf,
                    /*c_mod=*/(short)0, c, /*reuse_a=*/false, /*reuse_b=*/false);
            const int slot = t % DMAXC;
            #pragma unroll
            for (int r = 0; r < 8; ++r) {
                const int m = r + (hi ? 8 : 0);     // C/D: VGPR r -> M=r / M=8+r
                const int n = mr + n0;
                if (n < P_N) {
                    // each row of E has an exact 1.0 entry -> c[r] >= exp(-eta) > 0
                    const float beta = MS[m] + __logf(c[r]);
                    Q[(slot * TB + m) * P_N + n] = beta - US[m * P_N + n];
                }
            }
        }
        __syncthreads();
    }
}

extern "C" void kernel_launch(void* const* d_in, const int* in_sizes, int n_in,
                              void* d_out, int out_size, void* d_ws, size_t ws_size,
                              hipStream_t stream) {
    const float*         theta = (const float*)d_in[0];          // [512,24,288] f32
    const float*         dur   = (const float*)d_in[1];          // [24,96] f32
    const unsigned char* mask  = (const unsigned char*)d_in[2];  // [288,24] bool
    float*               out   = (float*)d_out;                  // [512] f32
    dim3 grid(B_TOT / TB);   // 32 workgroups, 16 batches each
    dim3 block(256);         // 8 wave32 waves
    hipLaunchKernelGGL(smcrf_fwd_kernel, grid, block, 0, stream,
                       theta, dur, mask, out);
}
